// MultiHeadSelfAttentionLayer_8890582302991
// MI455X (gfx1250) — matrix operand order repaired and checked by reading, not compiled, not run on hardware
//
#include <hip/hip_runtime.h>
#include <math.h>
#include <stdint.h>

#define BB 2
#define SS 2048
#define DD 1024
#define HH 16
#define HDIM 64
#define MM (BB*SS)   // 4096 rows of X

typedef _Float16 f16;
typedef __attribute__((ext_vector_type(16))) _Float16 v16h;
typedef __attribute__((ext_vector_type(8)))  _Float16 v8h;
typedef __attribute__((ext_vector_type(4)))  _Float16 v4h;
typedef __attribute__((ext_vector_type(8)))  float    v8f;
typedef __attribute__((ext_vector_type(4)))  int      v4i;

#if defined(__gfx1250__) && __has_builtin(__builtin_amdgcn_global_load_async_to_lds_b128)
#define HAVE_ASYNC_LDS 1
#else
#define HAVE_ASYNC_LDS 0
#endif

typedef __attribute__((address_space(1))) v4i gv4i;   // global int4
typedef __attribute__((address_space(3))) v4i lv4i;   // LDS int4

// 16-byte global -> LDS copy. Async (ASYNCcnt-tracked, no VGPR round-trip) on
// gfx1250; falls back to load+ds_store otherwise.
static __device__ __forceinline__ void cp16_g2l(const f16* g, f16* l) {
#if HAVE_ASYNC_LDS
  __builtin_amdgcn_global_load_async_to_lds_b128(
      (gv4i*)(uintptr_t)g,
      (lv4i*)(uint32_t)(uintptr_t)l,   // low 32 bits of generic ptr = LDS offset
      0, 0);
#else
  *(v8h*)l = *(const v8h*)g;
#endif
}

template <int N>
static __device__ __forceinline__ void wait_async() {
#if HAVE_ASYNC_LDS
#if __has_builtin(__builtin_amdgcn_s_wait_asynccnt)
  __builtin_amdgcn_s_wait_asynccnt(N);
#else
  if constexpr (N == 0) asm volatile("s_wait_asynccnt 0x0" ::: "memory");
  else                  asm volatile("s_wait_asynccnt 0x6" ::: "memory");
#endif
#endif
}

static __device__ __forceinline__ v16h cat8(v8h lo, v8h hi) {
  return __builtin_shufflevector(lo, hi, 0,1,2,3,4,5,6,7,8,9,10,11,12,13,14,15);
}

static __device__ __forceinline__ v8f wmma16(v16h a, v16h b, v8f c) {
  // D = A(16x32 f16) * B(32x16 f16) + C(16x16 f32)
  return __builtin_amdgcn_wmma_f32_16x16x32_f16(false, a, false, b, (short)0, c, false, false);
}

// A-operand gather (16x32 f16) from row-major [row][stride] f16 data.
// lane l holds row (l&15); halves = K[(l>>4)*8 .. +8) and K[16+(l>>4)*8 .. +8)
static __device__ __forceinline__ v16h load_A(const f16* base, int stride, int row, int kofs, int lane) {
  const f16* p = base + (size_t)row * stride + kofs + ((lane >> 4) << 3);
  v8h lo = *(const v8h*)p;
  v8h hi = *(const v8h*)(p + 16);
  return cat8(lo, hi);
}

// B-operand gather (32x16 f16) from "B^T row-major" data: lane l holds column
// n=(l&15) (a row of B^T); halves = K[(l>>4)*8 .. +8) and K[16+(l>>4)*8 .. +8).
static __device__ __forceinline__ v16h load_B(const f16* base, int stride, int nrow, int kofs, int lane) {
  const f16* p = base + (size_t)nrow * stride + kofs + ((lane >> 4) << 3);
  v8h lo = *(const v8h*)p;
  v8h hi = *(const v8h*)(p + 16);
  return cat8(lo, hi);
}

// ---------------------------------------------------------------- prep kernels
__global__ __launch_bounds__(256) void convert_x(const float* __restrict__ x,
                                                 f16* __restrict__ Xh) {
  int i = (blockIdx.x * 256 + threadIdx.x) * 4;
  float4 v = *(const float4*)(x + i);
  v4h h; h.x = (f16)v.x; h.y = (f16)v.y; h.z = (f16)v.z; h.w = (f16)v.w;
  *(v4h*)(Xh + i) = h;
}

// W [D][D] f32 row-major -> WT [n][k] f16 (WT[n][k] = W[k][n]); z picks q/k/v
__global__ __launch_bounds__(256) void transpose_w(const float* __restrict__ Wq,
                                                   const float* __restrict__ Wk,
                                                   const float* __restrict__ Wv,
                                                   f16* __restrict__ WT_all) {
  __shared__ float tile[32][33];
  const int z = blockIdx.z;
  const float* W = (z == 0) ? Wq : (z == 1) ? Wk : Wv;
  f16* WT = WT_all + (size_t)z * DD * DD;
  int x0 = blockIdx.x * 32 + threadIdx.x;   // col n of W
  int y0 = blockIdx.y * 32 + threadIdx.y;   // row k of W
  #pragma unroll
  for (int i = 0; i < 32; i += 8)
    tile[threadIdx.y + i][threadIdx.x] = W[(size_t)(y0 + i) * DD + x0];
  __syncthreads();
  int x1 = blockIdx.y * 32 + threadIdx.x;   // col k of WT
  int y1 = blockIdx.x * 32 + threadIdx.y;   // row n of WT
  #pragma unroll
  for (int i = 0; i < 32; i += 8)
    WT[(size_t)(y1 + i) * DD + x1] = (f16)tile[threadIdx.x][threadIdx.y + i];
}

// ---------------------------------------------------------------- QKV GEMM
// C[m][n] = X[m][:] . W[:][n] + bias[n], f16 WMMA, f32 accumulate.
// 256 threads = 8 waves, 64(M) x 128(N) tile, K-step 64, double-buffered LDS
// filled with async global->LDS copies (ASYNCcnt) so step i+1's copies overlap
// step i's WMMAs.
// z==0 -> Q [B,H,S,HD], z==1 -> K [B,H,S,HD], z==2 -> V^T [B,H,HD,S]
__global__ __launch_bounds__(256) void qkv_gemm(const f16* __restrict__ Xh,
                                                const f16* __restrict__ WT_all,
                                                const float* __restrict__ bq,
                                                const float* __restrict__ bk,
                                                const float* __restrict__ bv,
                                                f16* __restrict__ Q,
                                                f16* __restrict__ Kh,
                                                f16* __restrict__ Vt) {
  __shared__ f16 As[2][64 * 64];    // 16 KB
  __shared__ f16 Bs[2][128 * 64];   // 32 KB
  const int z = blockIdx.z;
  const f16* WT = WT_all + (size_t)z * DD * DD;
  const float* bias = (z == 0) ? bq : (z == 1) ? bk : bv;
  f16* outp = (z == 0) ? Q : (z == 1) ? Kh : Vt;

  const int t = threadIdx.x;
  const int lane = t & 31, w = t >> 5;
  const int mw = (w & 3) * 16, nw = (w >> 2) * 64;
  const int mBase = blockIdx.y * 64;
  const int nBase = blockIdx.x * 128;

  v8f acc0 = {}, acc1 = {}, acc2 = {}, acc3 = {};

  // cooperative staging: A 16 halves/thread (2 x b128), B 32 halves/thread (4 x b128)
  const int arow = t >> 2, acol = (t & 3) * 16;
  const int brow = t >> 1, bcol = (t & 1) * 32;

  auto stage = [&](int buf, int kk) {
    const f16* ga = &Xh[(size_t)(mBase + arow) * DD + kk + acol];
    f16*       la = &As[buf][arow * 64 + acol];
    cp16_g2l(ga,      la);
    cp16_g2l(ga + 8,  la + 8);
    const f16* gb = &WT[(size_t)(nBase + brow) * DD + kk + bcol];
    f16*       lb = &Bs[buf][brow * 64 + bcol];
    cp16_g2l(gb,      lb);
    cp16_g2l(gb + 8,  lb + 8);
    cp16_g2l(gb + 16, lb + 16);
    cp16_g2l(gb + 24, lb + 24);
  };

  stage(0, 0);
  const int NITER = DD / 64;   // 16
  for (int i = 0; i < NITER; ++i) {
    if (i + 1 < NITER) { stage((i + 1) & 1, (i + 1) * 64); wait_async<6>(); }
    else               { wait_async<0>(); }
    __syncthreads();                       // all waves' tiles for step i landed
    const f16* Ab = &As[i & 1][0];
    const f16* Bb = &Bs[i & 1][0];
    #pragma unroll
    for (int kc = 0; kc < 64; kc += 32) {
      v16h a  = load_A(Ab, 64, mw + (lane & 15), kc, lane);
      v16h b0 = load_B(Bb, 64, nw +  0 + (lane & 15), kc, lane);
      v16h b1 = load_B(Bb, 64, nw + 16 + (lane & 15), kc, lane);
      v16h b2 = load_B(Bb, 64, nw + 32 + (lane & 15), kc, lane);
      v16h b3 = load_B(Bb, 64, nw + 48 + (lane & 15), kc, lane);
      acc0 = wmma16(a, b0, acc0);
      acc1 = wmma16(a, b1, acc1);
      acc2 = wmma16(a, b2, acc2);
      acc3 = wmma16(a, b3, acc3);
    }
    __syncthreads();                       // step i reads done; buffer reusable
  }

  // Epilogue. C layout: VGPR r, lanes 0-15 -> M=r, N=lane; lanes 16-31 -> M=8+r.
  const int mrow0 = mBase + mw + ((lane >> 4) << 3);
  const int ncol0 = nBase + nw + (lane & 15);
  #pragma unroll
  for (int r = 0; r < 8; ++r) {
    int m = mrow0 + r;
    int bI = m >> 11, s = m & (SS - 1);
    float vals[4] = {acc0[r], acc1[r], acc2[r], acc3[r]};
    #pragma unroll
    for (int tt = 0; tt < 4; ++tt) {
      int n = ncol0 + tt * 16;
      float v = vals[tt] + bias[n];
      int h = n >> 6, hd = n & 63;
      size_t idx = (z == 2)
          ? ((size_t)((bI * HH + h) * HDIM + hd)) * SS + s   // V^T
          : ((size_t)((bI * HH + h) * SS + s)) * HDIM + hd;  // Q, K
      outp[idx] = (f16)v;
    }
  }
}

// ---------------------------------------------------------------- attention
// Flash-style: each wave owns 16 query rows; streams 32 keys/iteration.
__global__ __launch_bounds__(256) void attention(const f16* __restrict__ Q,
                                                 const f16* __restrict__ Kh,
                                                 const f16* __restrict__ Vt,
                                                 const float* __restrict__ x,
                                                 float* __restrict__ out) {
  __shared__ f16 Plds[8][16 * 32];
  const int lane = threadIdx.x & 31, w = threadIdx.x >> 5;
  const int b = blockIdx.z, h = blockIdx.y;
  const int qbase = blockIdx.x * 128 + w * 16;

  const f16* Qb = Q  + (size_t)(b * HH + h) * SS * HDIM;
  const f16* Kb = Kh + (size_t)(b * HH + h) * SS * HDIM;
  const f16* Vb = Vt + (size_t)(b * HH + h) * HDIM * SS;

  // Q A-operands (16 rows x 64), pre-scaled by 1/sqrt(HD) = 1/8 (exact in f16)
  v16h qa0 = load_A(Qb, HDIM, qbase + (lane & 15),  0, lane) * (_Float16)0.125f;
  v16h qa1 = load_A(Qb, HDIM, qbase + (lane & 15), 32, lane) * (_Float16)0.125f;

  v8f o0 = {}, o1 = {}, o2 = {}, o3 = {};
  float m_r[8], l_r[8];
  #pragma unroll
  for (int r = 0; r < 8; ++r) { m_r[r] = -1e30f; l_r[r] = 0.0f; }

  f16* P = &Plds[w][0];
  const v8f zero = {};

  for (int kb = 0; kb < SS; kb += 32) {
    // scores: 16 q-rows x 32 keys, two 16x16 C-tiles, each HD=64 = 2 WMMAs
    v16h k0lo = load_B(Kb, HDIM, kb +      (lane & 15),  0, lane);
    v16h k0hi = load_B(Kb, HDIM, kb +      (lane & 15), 32, lane);
    v16h k1lo = load_B(Kb, HDIM, kb + 16 + (lane & 15),  0, lane);
    v16h k1hi = load_B(Kb, HDIM, kb + 16 + (lane & 15), 32, lane);
    v8f s0 = wmma16(qa0, k0lo, zero); s0 = wmma16(qa1, k0hi, s0);
    v8f s1 = wmma16(qa0, k1lo, zero); s1 = wmma16(qa1, k1hi, s1);

    __syncthreads();  // P region free for reuse (prior loads done)
    #pragma unroll
    for (int r = 0; r < 8; ++r) {
      // row reductions across 16-lane half-groups (C layout: fixed row per r)
      float rmax = fmaxf(s0[r], s1[r]);
      for (int off = 8; off; off >>= 1) rmax = fmaxf(rmax, __shfl_xor(rmax, off, 32));
      float newm  = fmaxf(m_r[r], rmax);
      float scale = __expf(m_r[r] - newm);
      float p0 = __expf(s0[r] - newm), p1 = __expf(s1[r] - newm);
      float rsum = p0 + p1;
      for (int off = 8; off; off >>= 1) rsum += __shfl_xor(rsum, off, 32);
      l_r[r] = l_r[r] * scale + rsum;
      m_r[r] = newm;
      o0[r] *= scale; o1[r] *= scale; o2[r] *= scale; o3[r] *= scale;
      // stash P (C layout -> LDS row-major 16x32)
      int M = r + ((lane >> 4) << 3);
      P[M * 32 +      (lane & 15)] = (f16)p0;
      P[M * 32 + 16 + (lane & 15)] = (f16)p1;
    }
    __syncthreads();

    // O += P(16x32) * V(32x64): A from LDS, B from V^T (contiguous in s)
    v16h pa  = load_A(P, 32, lane & 15, 0, lane);
    v16h vb0 = load_B(Vb, SS,  0 + (lane & 15), kb, lane);
    v16h vb1 = load_B(Vb, SS, 16 + (lane & 15), kb, lane);
    v16h vb2 = load_B(Vb, SS, 32 + (lane & 15), kb, lane);
    v16h vb3 = load_B(Vb, SS, 48 + (lane & 15), kb, lane);
    o0 = wmma16(pa, vb0, o0);
    o1 = wmma16(pa, vb1, o1);
    o2 = wmma16(pa, vb2, o2);
    o3 = wmma16(pa, vb3, o3);
  }

  // epilogue: normalize, add residual, write f32
  const int srow0 = qbase + ((lane >> 4) << 3);
  #pragma unroll
  for (int r = 0; r < 8; ++r) {
    int s = srow0 + r;
    float inv = 1.0f / l_r[r];
    float vals[4] = {o0[r], o1[r], o2[r], o3[r]};
    #pragma unroll
    for (int tt = 0; tt < 4; ++tt) {
      int n = h * HDIM + tt * 16 + (lane & 15);
      size_t idx = ((size_t)(b * SS + s)) * DD + n;
      out[idx] = vals[tt] * inv + x[idx];
    }
  }
}

// ---------------------------------------------------------------- launch
extern "C" void kernel_launch(void* const* d_in, const int* in_sizes, int n_in,
                              void* d_out, int out_size, void* d_ws, size_t ws_size,
                              hipStream_t stream) {
  (void)in_sizes; (void)n_in; (void)out_size; (void)ws_size;
  const float* x  = (const float*)d_in[0];
  const float* Wq = (const float*)d_in[1];
  const float* bq = (const float*)d_in[2];
  const float* Wk = (const float*)d_in[3];
  const float* bk = (const float*)d_in[4];
  const float* Wv = (const float*)d_in[5];
  const float* bv = (const float*)d_in[6];
  float* out = (float*)d_out;

  char* ws = (char*)d_ws;
  const size_t MB = 1024 * 1024;
  f16* Xh = (f16*)(ws);              //  8 MB: X as f16 [4096][1024]
  f16* WT = (f16*)(ws +  8 * MB);    //  6 MB: Wq^T,Wk^T,Wv^T f16 [3][1024][1024]
  f16* Q  = (f16*)(ws + 14 * MB);    //  8 MB: [B,H,S,HD]
  f16* Kh = (f16*)(ws + 22 * MB);    //  8 MB: [B,H,S,HD]
  f16* Vt = (f16*)(ws + 30 * MB);    //  8 MB: [B,H,HD,S]

  convert_x  <<<dim3((MM * DD) / (4 * 256)), 256, 0, stream>>>(x, Xh);
  transpose_w<<<dim3(32, 32, 3), dim3(32, 8), 0, stream>>>(Wq, Wk, Wv, WT);
  qkv_gemm   <<<dim3(DD / 128, MM / 64, 3), 256, 0, stream>>>(Xh, WT, bq, bk, bv, Q, Kh, Vt);
  attention  <<<dim3(SS / 128, HH, BB), 256, 0, stream>>>(Q, Kh, Vt, x, out);
}
